// MixHopLayer_47107201303137
// MI455X (gfx1250) — matrix operand order repaired and checked by reading, not compiled, not run on hardware
//
#include <hip/hip_runtime.h>
#include <cstdint>

typedef __attribute__((ext_vector_type(16))) _Float16 v16h;
typedef __attribute__((ext_vector_type(8)))  _Float16 v8h;
typedef __attribute__((ext_vector_type(8)))  float    v8f;
typedef __attribute__((ext_vector_type(4)))  float    v4f;

#define F 128

// ---------------- utility kernels ----------------

__global__ void fill0_i32(int* p, int n) {
  int i = blockIdx.x * blockDim.x + threadIdx.x;
  if (i < n) p[i] = 0;
}

__global__ void cvt_w_f16(const float* __restrict__ W, _Float16* __restrict__ Wh, int n) {
  int i = blockIdx.x * blockDim.x + threadIdx.x;
  if (i < n) Wh[i] = (_Float16)W[i];
}

// ---------------- GEMM: y_j = x @ W[j]^T + b[j], via v_wmma_f32_16x16x32_f16 ----------------
// One block = 8 waves = one 16-row tile; wave w owns column tile w (16 cols).
// Each wave builds 4 A fragments (K = 0..127 in 4 steps of 32), reused across j=0,1,2.

__global__ __launch_bounds__(256) void gemm_wmma(
    const float* __restrict__ x, const _Float16* __restrict__ Wh,
    const float* __restrict__ bias,
    float* __restrict__ d0, float* __restrict__ d1, float* __restrict__ d2,
    int N)
{
  const int lane = threadIdx.x & 31;
  const int wv   = threadIdx.x >> 5;   // column tile 0..7
  const int rowt = blockIdx.x;
  const int hm   = lane >> 4;          // lane-half select
  const int l15  = lane & 15;
  const bool full_tile = (rowt * 16 + 16) <= N;

  int m = rowt * 16 + l15;
  if (m >= N) m = N - 1;               // clamp loads; stores guarded below
  const int n = wv * 16 + l15;

  // A fragments: 16-bit A 16x32 layout.
  // lanes 0-15: elems 0..7 -> K=k0+0..7,  elems 8..15 -> K=k0+16..23
  // lanes16-31: elems 0..7 -> K=k0+8..15, elems 8..15 -> K=k0+24..31
  v16h afrag[4];
#pragma unroll
  for (int s = 0; s < 4; ++s) {
    const int k0 = s * 32;
    const float* pa = x + (size_t)m * F + k0 + hm * 8;
    v4f f0 = *(const v4f*)(pa);
    v4f f1 = *(const v4f*)(pa + 4);
    v4f f2 = *(const v4f*)(pa + 16);
    v4f f3 = *(const v4f*)(pa + 20);
    v16h a;
#pragma unroll
    for (int i = 0; i < 4; ++i) {
      a[i]      = (_Float16)f0[i];
      a[4 + i]  = (_Float16)f1[i];
      a[8 + i]  = (_Float16)f2[i];
      a[12 + i] = (_Float16)f3[i];
    }
    afrag[s] = a;
  }

  float* dsts[3] = { d0, d1, d2 };
  const int ldds[3] = { 3 * F, F, 3 * F };
  const int mrow0 = rowt * 16 + hm * 8;   // C/D layout: lanes>=16 hold M=8..15

#pragma unroll
  for (int j = 0; j < 3; ++j) {
    v8f c = {};
#pragma unroll
    for (int s = 0; s < 4; ++s) {
      const int k0 = s * 32;
      // B layout: lanes 0-15 hold K=k0+0..7,16..23; lanes 16-31 hold K=k0+8..15,24..31 (matches A)
      const _Float16* pb = Wh + ((size_t)j * F + n) * F + k0 + hm * 8;
      v8h b0 = *(const v8h*)(pb);
      v8h b1 = *(const v8h*)(pb + 16);
      v16h bf;
#pragma unroll
      for (int i = 0; i < 8; ++i) { bf[i] = b0[i]; bf[8 + i] = b1[i]; }
      c = __builtin_amdgcn_wmma_f32_16x16x32_f16(
            false, afrag[s], false, bf, (short)0, c, false, false);
    }
    const float bv = bias[j * F + n];
    const int ldd = ldds[j];
    float* __restrict__ dptr = dsts[j] + (size_t)mrow0 * ldd + n;
    if (full_tile) {
      // branchless: 8 stores at immediate offsets v*ldd*4 (<= 10752 B, fits IOFFSET)
#pragma unroll
      for (int v = 0; v < 8; ++v) dptr[(size_t)v * ldd] = c[v] + bv;
    } else {
#pragma unroll
      for (int v = 0; v < 8; ++v) {
        if (mrow0 + v < N) dptr[(size_t)v * ldd] = c[v] + bv;
      }
    }
  }
}

// ---------------- CSR build ----------------

__global__ void hist_rows(const int* __restrict__ row, int* __restrict__ deg, int E) {
  int e = blockIdx.x * blockDim.x + threadIdx.x;
  if (e < E) atomicAdd(&deg[row[e]], 1);
}

__global__ void scan_excl(const int* __restrict__ deg, int* __restrict__ offs, int n) {
  __shared__ int buf[256];
  __shared__ int carry;
  const int t = threadIdx.x;
  if (t == 0) carry = 0;
  __syncthreads();
  for (int base = 0; base < n; base += 256) {
    const int v = (base + t < n) ? deg[base + t] : 0;
    buf[t] = v;
    __syncthreads();
    for (int d = 1; d < 256; d <<= 1) {
      const int add = (t >= d) ? buf[t - d] : 0;
      __syncthreads();
      buf[t] += add;
      __syncthreads();
    }
    if (base + t < n) offs[base + t] = carry + buf[t] - v;   // exclusive
    __syncthreads();
    if (t == 255) carry += buf[255];
    __syncthreads();
  }
  if (t == 0) offs[n] = carry;
}

__global__ void scatter_edges(const int* __restrict__ row, const int* __restrict__ offs,
                              int* __restrict__ cur, int* __restrict__ eid, int E) {
  int e = blockIdx.x * blockDim.x + threadIdx.x;
  if (e < E) {
    const int r = row[e];
    const int p = offs[r] + atomicAdd(&cur[r], 1);
    eid[p] = e;
  }
}

// canonical in-row order -> deterministic fp summation order (rows avg deg 16)
__global__ void sort_rows(const int* __restrict__ offs, int* __restrict__ eid, int N) {
  int i = blockIdx.x * blockDim.x + threadIdx.x;
  if (i >= N) return;
  const int s = offs[i], e = offs[i + 1];
  for (int a = s + 1; a < e; ++a) {
    const int key = eid[a];
    int b = a - 1;
    while (b >= s && eid[b] > key) { eid[b + 1] = eid[b]; --b; }
    eid[b + 1] = key;
  }
}

// ---------------- SPMM gather: one wave per output row, lane owns 4 columns ----------------

__global__ __launch_bounds__(256) void spmm_csr(
    const int* __restrict__ offs, const int* __restrict__ eid,
    const int* __restrict__ col,  const float* __restrict__ w,
    const float* __restrict__ src, int lds_src,
    float* __restrict__ dst, int ldd, int N)
{
  const int wid  = (blockIdx.x * blockDim.x + threadIdx.x) >> 5;
  const int lane = threadIdx.x & 31;
  if (wid >= N) return;
  const int e0 = offs[wid], e1 = offs[wid + 1];
  const int base = lane * 4;
  v4f acc = { 0.f, 0.f, 0.f, 0.f };
  for (int e = e0; e < e1; ++e) {
    const int   ed  = eid[e];
    const float wgt = w[ed];
    const int   c   = col[ed];
    const v4f v = *(const v4f*)(src + (size_t)c * lds_src + base);
    acc.x += wgt * v.x;
    acc.y += wgt * v.y;
    acc.z += wgt * v.z;
    acc.w += wgt * v.w;
  }
  *(v4f*)(dst + (size_t)wid * ldd + base) = acc;
}

// ---------------- launch ----------------

extern "C" void kernel_launch(void* const* d_in, const int* in_sizes, int n_in,
                              void* d_out, int out_size, void* d_ws, size_t ws_size,
                              hipStream_t stream) {
  const float* x    = (const float*)d_in[0];
  const float* ew   = (const float*)d_in[1];
  const float* W    = (const float*)d_in[2];
  const float* bias = (const float*)d_in[3];
  const int*   row  = (const int*)d_in[4];
  const int*   col  = (const int*)d_in[5];

  const int N = in_sizes[0] / F;
  const int E = in_sizes[1];

  float* out = (float*)d_out;

  // workspace carve (~30 MB)
  uint8_t* p = (uint8_t*)d_ws;
  auto take = [&](size_t bytes) -> void* {
    void* r = (void*)p;
    p += (bytes + 255) & ~(size_t)255;
    return r;
  };
  float*    tmpA = (float*)take((size_t)N * F * sizeof(float));
  _Float16* Wh   = (_Float16*)take((size_t)3 * F * F * sizeof(_Float16));
  int*      deg  = (int*)take((size_t)N * sizeof(int));
  int*      offs = (int*)take((size_t)(N + 1) * sizeof(int));
  int*      cur  = (int*)take((size_t)N * sizeof(int));
  int*      eid  = (int*)take((size_t)E * sizeof(int));

  const int TB = 256;
  fill0_i32<<<(N + TB - 1) / TB, TB, 0, stream>>>(deg, N);
  fill0_i32<<<(N + TB - 1) / TB, TB, 0, stream>>>(cur, N);
  cvt_w_f16<<<(3 * F * F + TB - 1) / TB, TB, 0, stream>>>(W, Wh, 3 * F * F);

  // y0 -> out[:,0:128]; y1 -> tmpA; y2 -> out[:,256:384]
  const int rowtiles = (N + 15) / 16;
  gemm_wmma<<<rowtiles, 256, 0, stream>>>(x, Wh, bias, out, tmpA, out + 2 * F, N);

  // CSR build (reused by all 3 spmm passes)
  hist_rows<<<(E + TB - 1) / TB, TB, 0, stream>>>(row, deg, E);
  scan_excl<<<1, 256, 0, stream>>>(deg, offs, N);
  scatter_edges<<<(E + TB - 1) / TB, TB, 0, stream>>>(row, offs, cur, eid, E);
  sort_rows<<<(N + TB - 1) / TB, TB, 0, stream>>>(offs, eid, N);

  // spmm passes: wave per row
  const int spmm_blocks = (N * 32 + TB - 1) / TB;
  // hop1: A @ y1 -> out[:,128:256]
  spmm_csr<<<spmm_blocks, TB, 0, stream>>>(offs, eid, col, ew, tmpA, F, out + F, 3 * F, N);
  // hop2 step 1: A @ y2 -> tmpA
  spmm_csr<<<spmm_blocks, TB, 0, stream>>>(offs, eid, col, ew, out + 2 * F, 3 * F, tmpA, F, N);
  // hop2 step 2: A @ (A @ y2) -> out[:,256:384]
  spmm_csr<<<spmm_blocks, TB, 0, stream>>>(offs, eid, col, ew, tmpA, F, out + 2 * F, 3 * F, N);
}
